// graph_network_44263932952753
// MI455X (gfx1250) — matrix operand order repaired and checked by reading, not compiled, not run on hardware
//
#include <hip/hip_runtime.h>
#include <stdint.h>

#define NN 262144
#define EN 524288
#define IN_DIM 32
#define HD 128
#define PROP_STEPS 3

typedef __attribute__((ext_vector_type(16))) __bf16 v16bf;
typedef __attribute__((ext_vector_type(8)))  float  v8f;

union FragBF16 { v16bf v; uint4 q[2]; };

__device__ __forceinline__ unsigned short f2bf(float f) {
    unsigned u = __float_as_uint(f);
    unsigned r = u + 0x7FFFu + ((u >> 16) & 1u);   // round-to-nearest-even
    return (unsigned short)(r >> 16);
}
__device__ __forceinline__ unsigned pk2bf(float a, float b) {
    return (unsigned)f2bf(a) | ((unsigned)f2bf(b) << 16);
}
// Branch-free tanh: copysign((1-e)/(1+e), x), e = exp(-2|x|)  (native exp + rcp)
__device__ __forceinline__ float tanh_fast(float x) {
    float e = __expf(-2.0f * __builtin_fabsf(x));
    float r = (1.0f - e) * __builtin_amdgcn_rcpf(1.0f + e);
    return __builtin_copysignf(r, x);
}
__device__ __forceinline__ float sigmoid_fast(float x) {
    return __builtin_amdgcn_rcpf(1.0f + __expf(-x));
}

// A fragment: 16x32 bf16 tile from row-major LDS (ldk halfs per row).
// Lane<16: K in {0..7, 16..23}; lane>=16: K in {8..15, 24..31} -> two b128 LDS loads.
__device__ __forceinline__ v16bf load_a(const unsigned short* lds, int row0, int kbase, int ldk) {
    int lane = threadIdx.x & 31;
    int r    = row0 + (lane & 15);
    int koff = (lane >> 4) * 8;
    FragBF16 a;
    a.q[0] = *(const uint4*)&lds[r * ldk + kbase + koff];
    a.q[1] = *(const uint4*)&lds[r * ldk + kbase + 16 + koff];
    return a.v;
}

// B fragment from pre-packed global weights: 32 bytes contiguous per lane.
__device__ __forceinline__ v16bf load_b(const unsigned short* pb, int nb, int kb, int KB) {
    int lane = threadIdx.x & 31;
    const uint4* p = (const uint4*)&pb[(size_t)((nb * KB + kb) * 32 + lane) * 16];
    FragBF16 b;
    b.q[0] = p[0];
    b.q[1] = p[1];
    return b.v;
}

__device__ __forceinline__ v8f splat8(float x) {
    v8f r;
    #pragma unroll
    for (int i = 0; i < 8; ++i) r[i] = x;
    return r;
}

// Pack row-major f32 W[K x N] into WMMA-B lane order, bf16.
__global__ void pack_b_kernel(const float* __restrict__ W, unsigned short* __restrict__ PB,
                              int K, int N) {
    int idx = blockIdx.x * blockDim.x + threadIdx.x;
    int KB = K >> 5, NB = N >> 4;
    int total = NB * KB * 512;
    if (idx >= total) return;
    int e    = idx & 15;
    int lane = (idx >> 4) & 31;
    int t    = idx >> 9;
    int kb   = t % KB;
    int nb   = t / KB;
    int k = kb * 32 + ((e >> 3) << 4) + ((lane >> 4) << 3) + (e & 7);
    int n = nb * 16 + (lane & 15);
    PB[idx] = f2bf(W[(size_t)k * N + n]);
}

__global__ void zero_f32v4_kernel(float4* __restrict__ p, int n4) {
    int i = blockIdx.x * blockDim.x + threadIdx.x;
    if (i < n4) p[i] = make_float4(0.f, 0.f, 0.f, 0.f);
}

__global__ void count_kernel(const long long* __restrict__ dstIdx, float* __restrict__ cnt) {
    int e = blockIdx.x * blockDim.x + threadIdx.x;
    if (e < EN) atomicAdd(&cnt[(int)dstIdx[e]], 1.0f);
}

__global__ void inv_cnt_kernel(float* __restrict__ c) {
    int i = blockIdx.x * blockDim.x + threadIdx.x;
    if (i < NN) c[i] = 1.0f / fmaxf(c[i], 1.0f);
}

// h = tanh(x @ W_in + b_in) ; K = 32 (single WMMA chunk)
__global__ __launch_bounds__(256)
void input_mlp_kernel(const float* __restrict__ x, const unsigned short* __restrict__ pWin,
                      const float* __restrict__ b_in, float* __restrict__ h) {
    __shared__ __align__(16) unsigned short lx[128 * IN_DIM];
    int tile = blockIdx.x, tid = threadIdx.x;
    const float4* xg = (const float4*)(x + (size_t)tile * 128 * IN_DIM);
    for (int i = tid; i < 128 * IN_DIM / 4; i += 256) {
        float4 v = xg[i];
        uint2 p; p.x = pk2bf(v.x, v.y); p.y = pk2bf(v.z, v.w);
        *(uint2*)&lx[i * 4] = p;
    }
    __syncthreads();
    int wave = tid >> 5, lane = tid & 31;
    int row0 = wave * 16;
    v16bf a = load_a(lx, row0, 0, IN_DIM);
    #pragma unroll
    for (int nb = 0; nb < 8; ++nb) {
        int col   = nb * 16 + (lane & 15);
        v8f acc   = splat8(b_in[col]);
        v16bf b = load_b(pWin, nb, 0, 1);
        acc = __builtin_amdgcn_wmma_f32_16x16x32_bf16(false, a, false, b, (short)0, acc, false, false);
        int rbase = tile * 128 + row0 + ((lane >> 4) << 3);
        #pragma unroll
        for (int v = 0; v < 8; ++v)
            h[(size_t)(rbase + v) * HD + col] = tanh_fast(acc[v]);
    }
}

// m = tanh(h @ We1 + b1) @ We2 + b2, fully fused in LDS
__global__ __launch_bounds__(256)
void edge_mlp_kernel(const float* __restrict__ h,
                     const unsigned short* __restrict__ pWe1, const float* __restrict__ b1,
                     const unsigned short* __restrict__ pWe2, const float* __restrict__ b2,
                     float* __restrict__ m) {
    __shared__ __align__(16) unsigned short lh[128 * HD];
    __shared__ __align__(16) unsigned short lt[128 * HD];
    int tile = blockIdx.x, tid = threadIdx.x;
    const float4* hg = (const float4*)(h + (size_t)tile * 128 * HD);
    for (int i = tid; i < 128 * HD / 4; i += 256) {
        float4 v = hg[i];
        uint2 p; p.x = pk2bf(v.x, v.y); p.y = pk2bf(v.z, v.w);
        *(uint2*)&lh[i * 4] = p;
    }
    __syncthreads();
    int wave = tid >> 5, lane = tid & 31;
    int row0 = wave * 16;

    v16bf afr[4];
    #pragma unroll
    for (int kb = 0; kb < 4; ++kb) afr[kb] = load_a(lh, row0, kb * 32, HD);
    for (int nb = 0; nb < 8; ++nb) {
        int col = nb * 16 + (lane & 15);
        v8f acc = splat8(b1[col]);
        #pragma unroll
        for (int kb = 0; kb < 4; ++kb) {
            v16bf b = load_b(pWe1, nb, kb, 4);
            acc = __builtin_amdgcn_wmma_f32_16x16x32_bf16(false, afr[kb], false, b, (short)0, acc, false, false);
        }
        int rl = row0 + ((lane >> 4) << 3);
        #pragma unroll
        for (int v = 0; v < 8; ++v)
            lt[(rl + v) * HD + col] = f2bf(tanh_fast(acc[v]));
    }
    __syncthreads();
    #pragma unroll
    for (int kb = 0; kb < 4; ++kb) afr[kb] = load_a(lt, row0, kb * 32, HD);
    for (int nb = 0; nb < 8; ++nb) {
        int col = nb * 16 + (lane & 15);
        v8f acc = splat8(b2[col]);
        #pragma unroll
        for (int kb = 0; kb < 4; ++kb) {
            v16bf b = load_b(pWe2, nb, kb, 4);
            acc = __builtin_amdgcn_wmma_f32_16x16x32_bf16(false, afr[kb], false, b, (short)0, acc, false, false);
        }
        int rg = tile * 128 + row0 + ((lane >> 4) << 3);
        #pragma unroll
        for (int v = 0; v < 8; ++v)
            m[(size_t)(rg + v) * HD + col] = acc[v];
    }
}

// agg[dst] += m[src] ; 8 threads per edge, 16 columns each
__global__ __launch_bounds__(256)
void scatter_kernel(const float* __restrict__ m, const long long* __restrict__ src,
                    const long long* __restrict__ dst, float* __restrict__ agg) {
    int gid = blockIdx.x * blockDim.x + threadIdx.x;
    int e = gid >> 3;
    if (e >= EN) return;
    int c0 = (gid & 7) * 16;
    int s = (int)src[e], d = (int)dst[e];
    const float4* mp = (const float4*)(m + (size_t)s * HD + c0);
    float* ap = agg + (size_t)d * HD + c0;
    #pragma unroll
    for (int j = 0; j < 4; ++j) {
        float4 v = mp[j];
        atomicAdd(ap + j * 4 + 0, v.x);
        atomicAdd(ap + j * 4 + 1, v.y);
        atomicAdd(ap + j * 4 + 2, v.z);
        atomicAdd(ap + j * 4 + 3, v.w);
    }
}

// Fused GRU: gi = (agg*invc) @ W_ih, gh = h @ W_hh, then gate math; h updated in place
__global__ __launch_bounds__(256)
void gru_kernel(const float* __restrict__ agg, const float* __restrict__ invc,
                const unsigned short* __restrict__ pWih, const float* __restrict__ b_ih,
                const unsigned short* __restrict__ pWhh, const float* __restrict__ b_hh,
                float* __restrict__ h) {
    __shared__ __align__(16) unsigned short lm[128 * HD];
    __shared__ __align__(16) unsigned short lh[128 * HD];
    int tile = blockIdx.x, tid = threadIdx.x;
    const float4* ag = (const float4*)(agg + (size_t)tile * 128 * HD);
    const float4* hg = (const float4*)(h + (size_t)tile * 128 * HD);
    for (int i = tid; i < 128 * HD / 4; i += 256) {
        int row = i >> 5;                    // (i*4)/128
        float ic = invc[tile * 128 + row];
        float4 a = ag[i]; float4 hv = hg[i];
        uint2 pa; pa.x = pk2bf(a.x * ic, a.y * ic); pa.y = pk2bf(a.z * ic, a.w * ic);
        uint2 ph; ph.x = pk2bf(hv.x, hv.y);         ph.y = pk2bf(hv.z, hv.w);
        *(uint2*)&lm[i * 4] = pa;
        *(uint2*)&lh[i * 4] = ph;
    }
    __syncthreads();
    int wave = tid >> 5, lane = tid & 31;
    int row0 = wave * 16;
    v16bf am[4], ah[4];
    #pragma unroll
    for (int kb = 0; kb < 4; ++kb) {
        am[kb] = load_a(lm, row0, kb * 32, HD);
        ah[kb] = load_a(lh, row0, kb * 32, HD);
    }
    const int KB = 4;  // W_ih/W_hh: K=128, N=384 -> NB=24
    for (int nb = 0; nb < 8; ++nb) {
        int col = nb * 16 + (lane & 15);
        v8f ir  = splat8(b_ih[col]);
        v8f iz  = splat8(b_ih[col + HD]);
        v8f in_ = splat8(b_ih[col + 2 * HD]);
        v8f hr  = splat8(b_hh[col]);
        v8f hz  = splat8(b_hh[col + HD]);
        v8f hn  = splat8(b_hh[col + 2 * HD]);
        #pragma unroll
        for (int kb = 0; kb < 4; ++kb) {
            v16bf br = load_b(pWih, nb,      kb, KB);
            v16bf bz = load_b(pWih, nb + 8,  kb, KB);
            v16bf bn = load_b(pWih, nb + 16, kb, KB);
            ir  = __builtin_amdgcn_wmma_f32_16x16x32_bf16(false, am[kb], false, br, (short)0, ir,  false, false);
            iz  = __builtin_amdgcn_wmma_f32_16x16x32_bf16(false, am[kb], false, bz, (short)0, iz,  false, false);
            in_ = __builtin_amdgcn_wmma_f32_16x16x32_bf16(false, am[kb], false, bn, (short)0, in_, false, false);
            v16bf cr = load_b(pWhh, nb,      kb, KB);
            v16bf cz = load_b(pWhh, nb + 8,  kb, KB);
            v16bf cn = load_b(pWhh, nb + 16, kb, KB);
            hr = __builtin_amdgcn_wmma_f32_16x16x32_bf16(false, ah[kb], false, cr, (short)0, hr, false, false);
            hz = __builtin_amdgcn_wmma_f32_16x16x32_bf16(false, ah[kb], false, cz, (short)0, hz, false, false);
            hn = __builtin_amdgcn_wmma_f32_16x16x32_bf16(false, ah[kb], false, cn, (short)0, hn, false, false);
        }
        int rg = tile * 128 + row0 + ((lane >> 4) << 3);
        #pragma unroll
        for (int v = 0; v < 8; ++v) {
            float r = sigmoid_fast(ir[v] + hr[v]);
            float z = sigmoid_fast(iz[v] + hz[v]);
            float n = tanh_fast(in_[v] + r * hn[v]);
            size_t off = (size_t)(rg + v) * HD + col;
            float ho = h[off];
            h[off] = (1.0f - z) * n + z * ho;
        }
    }
}

// out = h @ W_out + b_out   (N=1 -> per-node dot product)
__global__ __launch_bounds__(256)
void out_kernel(const float* __restrict__ h, const float* __restrict__ Wout,
                const float* __restrict__ bout, float* __restrict__ out) {
    __shared__ float lw[HD];
    int tid = threadIdx.x;
    if (tid < HD) lw[tid] = Wout[tid];
    __syncthreads();
    int node = blockIdx.x * blockDim.x + tid;
    if (node >= NN) return;
    const float4* hp = (const float4*)(h + (size_t)node * HD);
    float s = 0.f;
    #pragma unroll 8
    for (int j = 0; j < HD / 4; ++j) {
        float4 v = hp[j];
        s += v.x * lw[j * 4] + v.y * lw[j * 4 + 1] + v.z * lw[j * 4 + 2] + v.w * lw[j * 4 + 3];
    }
    out[node] = s + bout[0];
}

extern "C" void kernel_launch(void* const* d_in, const int* in_sizes, int n_in,
                              void* d_out, int out_size, void* d_ws, size_t ws_size,
                              hipStream_t stream) {
    (void)in_sizes; (void)n_in; (void)out_size; (void)ws_size;
    const float*     x     = (const float*)d_in[0];
    const long long* ei    = (const long long*)d_in[1];   // int64 [2, EN]
    const float*     W_in  = (const float*)d_in[2];
    const float*     b_in  = (const float*)d_in[3];
    const float*     W_e1  = (const float*)d_in[4];
    const float*     b_e1  = (const float*)d_in[5];
    const float*     W_e2  = (const float*)d_in[6];
    const float*     b_e2  = (const float*)d_in[7];
    const float*     W_ih  = (const float*)d_in[8];
    const float*     b_ih  = (const float*)d_in[9];
    const float*     W_hh  = (const float*)d_in[10];
    const float*     b_hh  = (const float*)d_in[11];
    const float*     W_out = (const float*)d_in[12];
    const float*     b_out = (const float*)d_in[13];
    float* out = (float*)d_out;

    char* wp = (char*)d_ws;
    auto carve = [&](size_t bytes) { char* p = wp; wp += (bytes + 255) & ~(size_t)255; return p; };
    float* h    = (float*)carve((size_t)NN * HD * 4);
    float* m    = (float*)carve((size_t)NN * HD * 4);
    float* agg  = (float*)carve((size_t)NN * HD * 4);
    float* cnt  = (float*)carve((size_t)NN * 4);
    unsigned short* pWin = (unsigned short*)carve((size_t)8  * 1 * 512 * 2);
    unsigned short* pWe1 = (unsigned short*)carve((size_t)8  * 4 * 512 * 2);
    unsigned short* pWe2 = (unsigned short*)carve((size_t)8  * 4 * 512 * 2);
    unsigned short* pWih = (unsigned short*)carve((size_t)24 * 4 * 512 * 2);
    unsigned short* pWhh = (unsigned short*)carve((size_t)24 * 4 * 512 * 2);

    const long long* srcI = ei;       // in_node
    const long long* dstI = ei + EN;  // out_node

    // One-time weight packing into WMMA-B lane order (bf16)
    pack_b_kernel<<<(8 * 1 * 512 + 255) / 256, 256, 0, stream>>>(W_in, pWin, 32, 128);
    pack_b_kernel<<<(8 * 4 * 512 + 255) / 256, 256, 0, stream>>>(W_e1, pWe1, 128, 128);
    pack_b_kernel<<<(8 * 4 * 512 + 255) / 256, 256, 0, stream>>>(W_e2, pWe2, 128, 128);
    pack_b_kernel<<<(24 * 4 * 512 + 255) / 256, 256, 0, stream>>>(W_ih, pWih, 128, 384);
    pack_b_kernel<<<(24 * 4 * 512 + 255) / 256, 256, 0, stream>>>(W_hh, pWhh, 128, 384);

    // Degree counts -> 1/max(cnt,1) (constant across steps)
    zero_f32v4_kernel<<<(NN / 4 + 255) / 256, 256, 0, stream>>>((float4*)cnt, NN / 4);
    count_kernel<<<(EN + 255) / 256, 256, 0, stream>>>(dstI, cnt);
    inv_cnt_kernel<<<(NN + 255) / 256, 256, 0, stream>>>(cnt);

    // h = tanh(x @ W_in + b_in)
    input_mlp_kernel<<<NN / 128, 256, 0, stream>>>(x, pWin, b_in, h);

    for (int s = 0; s < PROP_STEPS; ++s) {
        edge_mlp_kernel<<<NN / 128, 256, 0, stream>>>(h, pWe1, b_e1, pWe2, b_e2, m);
        zero_f32v4_kernel<<<((size_t)NN * HD / 4 + 255) / 256, 256, 0, stream>>>((float4*)agg, NN * HD / 4);
        scatter_kernel<<<(EN * 8 + 255) / 256, 256, 0, stream>>>(m, srcI, dstI, agg);
        gru_kernel<<<NN / 128, 256, 0, stream>>>(agg, cnt, pWih, b_ih, pWhh, b_hh, h);
    }

    out_kernel<<<(NN + 255) / 256, 256, 0, stream>>>(h, W_out, b_out, out);
}
